// LayerBlock_89928025243752
// MI455X (gfx1250) — matrix operand order repaired and checked by reading, not compiled, not run on hardware
//
#include <hip/hip_runtime.h>
#include <hip/hip_bf16.h>

typedef _Float16 h16 __attribute__((ext_vector_type(16)));
typedef _Float16 h8  __attribute__((ext_vector_type(8)));
typedef float    f8  __attribute__((ext_vector_type(8)));

#define F8Z {0.f,0.f,0.f,0.f,0.f,0.f,0.f,0.f}
#define BN_SCALE 0.9999950000374997f   /* 1/sqrt(1+1e-5) */

__device__ __forceinline__ f8 wmma_f16(h16 a, h16 b, f8 c) {
  // emits v_wmma_f32_16x16x32_f16
  return __builtin_amdgcn_wmma_f32_16x16x32_f16(false, a, false, b, (short)0, c, false, false);
}

// A-fragment (16x32 f16, layout [row][k]): elems 0..7 = K k0..k0+7, 8..15 = K k0+16..k0+23
__device__ __forceinline__ h16 frag_ld_a(const _Float16* p) {
  union { h16 v; h8 h[2]; } u;
  u.h[0] = *(const h8*)(p);
  u.h[1] = *(const h8*)(p + 16);
  return u.v;
}
// B-fragment (32x16 f16, tile stored transposed [n][k]): elems 0..7 = K bk0..bk0+7, 8..15 = K bk0+16..bk0+23
__device__ __forceinline__ h16 frag_ld_b(const _Float16* p) {
  union { h16 v; h8 h[2]; } u;
  u.h[0] = *(const h8*)(p);
  u.h[1] = *(const h8*)(p + 16);
  return u.v;
}

// ---------------------------------------------------------------------------
// Tensor Data Mover: stage a 64x32 f16 W-tile (row stride K elems) into LDS.
// D# layout per CDNA5 ISA (group0 128b, group1 256b).
// Issued by one wave; completion via s_wait_tensorcnt.
// ---------------------------------------------------------------------------
#if __has_builtin(__builtin_amdgcn_tensor_load_to_lds)
#define HAVE_TDM 1
typedef unsigned int u32x4v __attribute__((ext_vector_type(4)));
typedef int          i32x4v __attribute__((ext_vector_type(4)));
typedef int          i32x8v __attribute__((ext_vector_type(8)));

__device__ __forceinline__ void tdm_load_w(const _Float16* gsrc, unsigned lds_byte, int K)
{
  const unsigned long long ga = (unsigned long long)(size_t)gsrc;
  u32x4v g0;
  g0[0] = 1u;                                        // count=1, user desc
  g0[1] = lds_byte;                                  // lds_addr
  g0[2] = (unsigned)(ga & 0xffffffffull);            // global_addr[31:0]
  g0[3] = (unsigned)((ga >> 32) & 0x01ffffffull)     // global_addr[56:32]
        | (2u << 30);                                // type=2 ("image")
  i32x8v g1;
  g1[0] = 1 << 16;                                   // wg_mask=0, data_size=1 (2 bytes)
  g1[1] = (K & 0xffff) << 16;                        // tensor_dim0[15:0]
  g1[2] = ((K >> 16) & 0xffff) | (64 << 16);         // tensor_dim0[31:16] | tensor_dim1[15:0]=64
  g1[3] = 32 << 16;                                  // tensor_dim1[31:16]=0 | tile_dim0=32
  g1[4] = 64;                                        // tile_dim1=64, tile_dim2=0
  g1[5] = K;                                         // tensor_dim0_stride[31:0]
  g1[6] = 0;                                         // stride hi | dim1_stride lo
  g1[7] = 0;
  const i32x4v z4 = {0, 0, 0, 0};
#if __clang_major__ >= 23
  const i32x8v z8 = {0, 0, 0, 0, 0, 0, 0, 0};
  __builtin_amdgcn_tensor_load_to_lds(g0, g1, z4, z4, z8, 0);
#else
  __builtin_amdgcn_tensor_load_to_lds(g0, g1, z4, z4, 0);
#endif
}
#endif

// ---------------------------------------------------------------------------
// Tiled WMMA GEMM (f16 W, f16 X, double-buffered LDS, TDM-staged W tiles):
//   OUT[b][m][n] = epi( W[m][k] * X[b][k][n] )
// EPI bits: 1=bias  2=relu  4=bn(g,beta)  8=residual-add(f32)
// M%64==0, N%64==0, K%32==0. grid (N/64, M/64, B), block 256.
// ---------------------------------------------------------------------------
__device__ __forceinline__ void stage_x(const _Float16* __restrict__ Xb,
                                        _Float16 (*Xt)[40],
                                        int n0, int kk, int N, int K, int t)
{
  const int xr = t >> 3, xc = (t & 7) * 8;
  const h8 xv = *(const h8*)&Xb[(size_t)(kk + xr) * N + n0 + xc];
  #pragma unroll
  for (int u = 0; u < 8; u++) Xt[xc + u][xr] = xv[u];
  if (kk + 32 < K)
    __builtin_prefetch(&Xb[(size_t)(kk + 32 + xr) * N + n0 + xc], 0, 0);
}

__device__ __forceinline__ void stage_w(const _Float16* __restrict__ W,
                                        _Float16 (*Wl)[32],
                                        int m0, int kk, int K, int t)
{
  const int wr = t >> 2, wc = (t & 3) * 8;
  *(h8*)&Wl[wr][wc] = *(const h8*)&W[(size_t)(m0 + wr) * K + kk + wc];
}

template<typename OT, int EPI>
__global__ void __launch_bounds__(256)
gemm_wmma(const _Float16* __restrict__ W, const _Float16* __restrict__ X,
          const float* __restrict__ bias, const float* __restrict__ bng,
          const float* __restrict__ bnb, const float* __restrict__ resid,
          OT* __restrict__ OUT, int M, int N, int K)
{
  __shared__ __align__(16) _Float16 Wl[2][64][32];   // TDM tile lands contiguous
  __shared__ __align__(16) _Float16 Xt[2][64][40];
  const int b  = blockIdx.z;
  const int m0 = blockIdx.y * 64, n0 = blockIdx.x * 64;
  const int t  = threadIdx.x, w = t >> 5, l = t & 31;
  const int i  = w >> 1, jb = (w & 1) * 2;
  const _Float16* Xb = X + (size_t)b * K * N;

  f8 acc0 = F8Z, acc1 = F8Z;

  const bool w0 = (t < 32);
#ifdef HAVE_TDM
  if (w0) tdm_load_w(&W[(size_t)m0 * K], (unsigned)(size_t)&Wl[0][0][0], K);
#else
  stage_w(W, Wl[0], m0, 0, K, t);
#endif
  stage_x(Xb, Xt[0], n0, 0, N, K, t);

  int pb = 0;
  const int ak0 = (l < 16) ? 0 : 8;
  const int bk0 = (l < 16) ? 0 : 8;
  const int ar  = i * 16 + (l & 15);
  const int bc  = jb * 16 + (l & 15);

  for (int kk = 0; kk < K; kk += 32) {
#ifdef HAVE_TDM
    if (w0) __builtin_amdgcn_s_wait_tensorcnt(0);    // current W tile arrived
#endif
    __syncthreads();
    if (kk + 32 < K) {
#ifdef HAVE_TDM
      if (w0) tdm_load_w(&W[(size_t)m0 * K + kk + 32],
                         (unsigned)(size_t)&Wl[pb ^ 1][0][0], K);
#else
      stage_w(W, Wl[pb ^ 1], m0, kk + 32, K, t);
#endif
      stage_x(Xb, Xt[pb ^ 1], n0, kk + 32, N, K, t);
    }
    const h16 af = frag_ld_a(&Wl[pb][ar][ak0]);
    const h16 b0 = frag_ld_b(&Xt[pb][bc][bk0]);
    const h16 b1 = frag_ld_b(&Xt[pb][bc + 16][bk0]);
    acc0 = wmma_f16(af, b0, acc0);
    acc1 = wmma_f16(af, b1, acc1);
    pb ^= 1;
  }

  #pragma unroll
  for (int jv = 0; jv < 8; jv++) {
    const int m = m0 + i * 16 + jv + ((l >= 16) ? 8 : 0);
    float v0 = acc0[jv], v1 = acc1[jv];
    if constexpr (EPI & 1) { const float bb = bias[m]; v0 += bb; v1 += bb; }
    if constexpr (EPI & 4) {
      const float sc = bng[m] * BN_SCALE, bt = bnb[m];
      v0 = v0 * sc + bt; v1 = v1 * sc + bt;
    }
    if constexpr (EPI & 2) { v0 = fmaxf(v0, 0.f); v1 = fmaxf(v1, 0.f); }
    const int nl = n0 + jb * 16 + (l & 15);
    const size_t oi = ((size_t)b * M + m) * N + nl;
    if constexpr (EPI & 8) { v0 += resid[oi]; v1 += resid[oi + 16]; }
    OUT[oi]      = (OT)v0;
    OUT[oi + 16] = (OT)v1;
  }
}

// ---------------------------------------------------------------------------
// KNN: pd = 2 x.xT - |x|^2 - |y|^2 via WMMA (f16 features), top-8 per row.
// grid (N/64, B), block 256
// ---------------------------------------------------------------------------
__global__ void __launch_bounds__(256)
knn_top8(const _Float16* __restrict__ dh, const float* __restrict__ sq,
         int* __restrict__ idx, int N)
{
  __shared__ __align__(16) _Float16 Xr[64][136];   // [row][c] (transposed tile)
  __shared__ __align__(16) _Float16 Xc[64][136];   // [col][c]
  __shared__ float Sc[64][65];
  const int b = blockIdx.y, r0 = blockIdx.x * 64;
  const _Float16* db = dh + (size_t)b * 128 * N;
  const int t = threadIdx.x, w = t >> 5, l = t & 31;
  const int i = w >> 1, jb = (w & 1) * 2;
  const int ak0 = (l < 16) ? 0 : 8;
  const int bk0 = (l < 16) ? 0 : 8;

  for (int e = t; e < 1024; e += 256) {
    const int c = e >> 3, rg = (e & 7) * 8;
    const h8 v = *(const h8*)&db[(size_t)c * N + r0 + rg];
    #pragma unroll
    for (int u = 0; u < 8; u++) Xr[rg + u][c] = v[u];
  }
  __syncthreads();

  float val[8]; int ind[8];
  #pragma unroll
  for (int u = 0; u < 8; u++) { val[u] = -1e30f; ind[u] = 0; }
  const float sqr = (t < 64) ? sq[b * N + r0 + t] : 0.f;

  for (int ct = 0; ct < N; ct += 64) {
    for (int e = t; e < 1024; e += 256) {
      const int c = e >> 3, rg = (e & 7) * 8;
      const h8 v = *(const h8*)&db[(size_t)c * N + ct + rg];
      #pragma unroll
      for (int u = 0; u < 8; u++) Xc[rg + u][c] = v[u];
    }
    __syncthreads();

    f8 a0 = F8Z, a1 = F8Z;
    #pragma unroll
    for (int kk = 0; kk < 128; kk += 32) {
      const h16 af = frag_ld_a(&Xr[i * 16 + (l & 15)][kk + ak0]);
      const h16 b0 = frag_ld_b(&Xc[jb * 16 + (l & 15)][kk + bk0]);
      const h16 b1 = frag_ld_b(&Xc[jb * 16 + 16 + (l & 15)][kk + bk0]);
      a0 = wmma_f16(af, b0, a0);
      a1 = wmma_f16(af, b1, a1);
    }
    #pragma unroll
    for (int jv = 0; jv < 8; jv++) {
      const int mrow = i * 16 + jv + ((l >= 16) ? 8 : 0);
      const int ncol = jb * 16 + (l & 15);
      Sc[mrow][ncol]      = 2.f * a0[jv];
      Sc[mrow][ncol + 16] = 2.f * a1[jv];
    }
    __syncthreads();
    if (t < 64) {
      for (int cc = 0; cc < 64; cc++) {
        const float v = Sc[t][cc] - sqr - sq[b * N + ct + cc];
        if (v > val[7]) {
          int p = 7;
          while (p > 0 && v > val[p - 1]) { val[p] = val[p - 1]; ind[p] = ind[p - 1]; --p; }
          val[p] = v; ind[p] = ct + cc;
        }
      }
    }
    __syncthreads();
  }
  if (t < 64) {
    #pragma unroll
    for (int u = 0; u < 8; u++) idx[((size_t)b * N + r0 + t) * 8 + u] = ind[u];
  }
}

// ---------------------------------------------------------------------------
// Flash attention, dh=32 per head. Q/K/V f16 [B][128][N].
// grid (N1/64, 4 heads, B), block 128 (4 waves; one 16-query tile per wave)
// ---------------------------------------------------------------------------
__global__ void __launch_bounds__(128)
attn_flash(const _Float16* __restrict__ Q, const _Float16* __restrict__ Kk,
           const _Float16* __restrict__ V, _Float16* __restrict__ O,
           int N1, int N2)
{
  __shared__ __align__(16) _Float16 Qs[64][40];    // [query][dh]
  __shared__ __align__(16) _Float16 Ks[64][40];    // [key][dh]
  __shared__ __align__(16) _Float16 Vs[32][72];    // [dh][key]
  __shared__ __align__(16) _Float16 Ps[4][16][64]; // per-wave P tile [row][key]
  const int b = blockIdx.z, hd = blockIdx.y;
  const int q0 = blockIdx.x * 64;
  const _Float16* Qb = Q  + ((size_t)b * 128 + hd * 32) * N1;
  const _Float16* Kb = Kk + ((size_t)b * 128 + hd * 32) * N2;
  const _Float16* Vb = V  + ((size_t)b * 128 + hd * 32) * N2;
  const int t = threadIdx.x, w = t >> 5, l = t & 31;
  const int ak0 = (l < 16) ? 0 : 8;
  const int bk0 = (l < 16) ? 0 : 8;

  for (int e = t; e < 256; e += 128) {
    const int c = e >> 3, rg = (e & 7) * 8;
    const h8 v = *(const h8*)&Qb[(size_t)c * N1 + q0 + rg];
    #pragma unroll
    for (int u = 0; u < 8; u++) Qs[rg + u][c] = v[u];
  }
  __syncthreads();

  const h16 qf = frag_ld_a(&Qs[w * 16 + (l & 15)][ak0]);

  f8 o0 = F8Z, o1 = F8Z;
  float rm[8], rs[8];
  #pragma unroll
  for (int jv = 0; jv < 8; jv++) { rm[jv] = -1e30f; rs[jv] = 0.f; }
  const float scale = 0.17677669529663687f; // 1/sqrt(32)

  for (int kt = 0; kt < N2; kt += 64) {
    for (int e = t; e < 256; e += 128) {
      const int c = e >> 3, rg = (e & 7) * 8;
      const h8 kv = *(const h8*)&Kb[(size_t)c * N2 + kt + rg];
      #pragma unroll
      for (int u = 0; u < 8; u++) Ks[rg + u][c] = kv[u];
      *(h8*)&Vs[c][rg] = *(const h8*)&Vb[(size_t)c * N2 + kt + rg];
    }
    __syncthreads();

    f8 s[4];
    #pragma unroll
    for (int sub = 0; sub < 4; sub++) {
      const h16 bf = frag_ld_b(&Ks[sub * 16 + (l & 15)][bk0]);
      f8 z = F8Z;
      s[sub] = wmma_f16(qf, bf, z);
    }

    #pragma unroll
    for (int jv = 0; jv < 8; jv++) {
      float mloc = -1e30f;
      #pragma unroll
      for (int sub = 0; sub < 4; sub++) {
        float v = s[sub][jv] * scale; s[sub][jv] = v;
        mloc = fmaxf(mloc, v);
      }
      for (int off = 1; off < 16; off <<= 1) mloc = fmaxf(mloc, __shfl_xor(mloc, off, 32));
      const float mnew = fmaxf(rm[jv], mloc);
      const float alpha = __expf(rm[jv] - mnew);
      float psum = 0.f;
      #pragma unroll
      for (int sub = 0; sub < 4; sub++) {
        float p = __expf(s[sub][jv] - mnew); s[sub][jv] = p; psum += p;
      }
      for (int off = 1; off < 16; off <<= 1) psum += __shfl_xor(psum, off, 32);
      rs[jv] = rs[jv] * alpha + psum;
      rm[jv] = mnew;
      o0[jv] *= alpha; o1[jv] *= alpha;
      const int prow = jv + ((l >= 16) ? 8 : 0);
      #pragma unroll
      for (int sub = 0; sub < 4; sub++) Ps[w][prow][sub * 16 + (l & 15)] = (_Float16)s[sub][jv];
    }

    // O += P * V   (A = Ps[w] 16x64 f16; two K-steps of 32 keys)
    #pragma unroll
    for (int ks = 0; ks < 2; ks++) {
      const h16 pf = frag_ld_a(&Ps[w][l & 15][ks * 32 + ak0]);
      #pragma unroll
      for (int j = 0; j < 2; j++) {
        const h16 vf = frag_ld_b(&Vs[j * 16 + (l & 15)][ks * 32 + bk0]);
        if (j == 0) o0 = wmma_f16(pf, vf, o0);
        else        o1 = wmma_f16(pf, vf, o1);
      }
    }
    __syncthreads();
  }

  #pragma unroll
  for (int jv = 0; jv < 8; jv++) {
    const int qrow = q0 + w * 16 + jv + ((l >= 16) ? 8 : 0);
    const float inv = 1.0f / rs[jv];
    const int d0 = l & 15;
    O[((size_t)b * 128 + hd * 32 + d0)      * N1 + qrow] = (_Float16)(o0[jv] * inv);
    O[((size_t)b * 128 + hd * 32 + 16 + d0) * N1 + qrow] = (_Float16)(o1[jv] * inv);
  }
}

// ---------------------------------------------------------------------------
// Small helpers
// ---------------------------------------------------------------------------
__global__ void f32_to_f16_k(const float* __restrict__ in, _Float16* __restrict__ out, size_t total)
{
  const size_t i = (size_t)blockIdx.x * 256 + threadIdx.x;
  if (i < total) out[i] = (_Float16)in[i];
}

__global__ void sqnorm_k(const float* __restrict__ d, float* __restrict__ sq, int N, int total)
{
  const int i = blockIdx.x * 256 + threadIdx.x;
  if (i >= total) return;
  const int b = i / N, n = i % N;
  const float* db = d + (size_t)b * 128 * N;
  float s = 0.f;
  for (int c = 0; c < 128; c++) { float v = db[(size_t)c * N + n]; s += v * v; }
  sq[i] = s;
}

// grid (N/32, B), block 256
__global__ void __launch_bounds__(256)
build_gf(const float* __restrict__ d, const int* __restrict__ idx,
         const float* __restrict__ gw, const float* __restrict__ gb,
         _Float16* __restrict__ gf, int N)
{
  __shared__ float sc[32][8];
  __shared__ float soft[32][8];
  __shared__ int   ji[32][8];
  const int b = blockIdx.y, n0 = blockIdx.x * 32;
  const float* db = d + (size_t)b * 128 * N;
  const int t = threadIdx.x;
  { const int nl = t >> 3, kk = t & 7;
    const int n = n0 + nl;
    const int j = idx[((size_t)b * N + n) * 8 + kk];
    ji[nl][kk] = j;
    float s = gb[0];
    for (int c = 0; c < 128; c++)
      s += (db[(size_t)c * N + n] - db[(size_t)c * N + j]) * gw[c];
    sc[nl][kk] = s;
  }
  __syncthreads();
  if (t < 32) {
    float mxv = -1e30f;
    #pragma unroll
    for (int k = 0; k < 8; k++) mxv = fmaxf(mxv, sc[t][k]);
    float e[8], sm = 0.f;
    #pragma unroll
    for (int k = 0; k < 8; k++) { e[k] = __expf(sc[t][k] - mxv); sm += e[k]; }
    #pragma unroll
    for (int k = 0; k < 8; k++) soft[t][k] = e[k] / sm;
  }
  __syncthreads();
  const size_t base = ((size_t)b * 256 + t) * (size_t)N * 8;
  for (int nl = 0; nl < 32; nl++) {
    const int n = n0 + nl;
    #pragma unroll
    for (int k = 0; k < 8; k++) {
      float v;
      if (t < 128) v = db[(size_t)t * N + n];
      else         v = soft[nl][k] * db[(size_t)(t - 128) * N + ji[nl][k]];
      gf[base + (size_t)n * 8 + k] = (_Float16)v;
    }
  }
}

__global__ void max_over_k(const _Float16* __restrict__ in, _Float16* __restrict__ out, size_t total)
{
  const size_t i = (size_t)blockIdx.x * 256 + threadIdx.x;
  if (i >= total) return;
  const _Float16* p = in + i * 8;
  float m = (float)p[0];
  #pragma unroll
  for (int k = 1; k < 8; k++) m = fmaxf(m, (float)p[k]);
  out[i] = (_Float16)m;
}

template<typename T0, typename T1>
__global__ void concat2_f16(const T0* __restrict__ a, const T1* __restrict__ b2,
                            _Float16* __restrict__ out, int C, int N, size_t total)
{
  const size_t i = (size_t)blockIdx.x * 256 + threadIdx.x;
  if (i >= total) return;
  const int n = (int)(i % N);
  const size_t r = i / N;
  const int ch = (int)(r % (2 * C));
  const int b = (int)(r / (2 * C));
  float v = (ch < C) ? (float)a[((size_t)b * C + ch) * N + n]
                     : (float)b2[((size_t)b * C + (ch - C)) * N + n];
  out[i] = (_Float16)v;
}

// in-place instance-norm (over N) + bn + relu;  grid = B*Cc
__global__ void __launch_bounds__(256)
inorm_bn_relu_k(float* __restrict__ x, const float* __restrict__ g,
                const float* __restrict__ be, int Cc, int N)
{
  const int bc = blockIdx.x, c = bc % Cc;
  float* xp = x + (size_t)bc * N;
  __shared__ float s1[256], s2[256];
  float a = 0.f, q = 0.f;
  for (int n = threadIdx.x; n < N; n += 256) { float v = xp[n]; a += v; q += v * v; }
  s1[threadIdx.x] = a; s2[threadIdx.x] = q;
  __syncthreads();
  for (int off = 128; off; off >>= 1) {
    if (threadIdx.x < off) { s1[threadIdx.x] += s1[threadIdx.x + off]; s2[threadIdx.x] += s2[threadIdx.x + off]; }
    __syncthreads();
  }
  const float mean = s1[0] / N;
  const float var  = s2[0] / N - mean * mean;
  const float inv  = rsqrtf(var + 1e-3f);
  const float scg  = g[c] * BN_SCALE, bt = be[c];
  for (int n = threadIdx.x; n < N; n += 256) {
    float v = (xp[n] - mean) * inv;
    xp[n] = fmaxf(v * scg + bt, 0.f);
  }
}

// 3x3 conv, SAME, 16x16 spatial, fused bn+relu. grid (Cout, B), block 256
__global__ void __launch_bounds__(256)
conv3x3_bnrelu(const float* __restrict__ x, const float* __restrict__ wt,
               const float* __restrict__ bias, const float* __restrict__ g,
               const float* __restrict__ be, float* __restrict__ y, int Cin)
{
  const int co = blockIdx.x, b = blockIdx.y;
  __shared__ float pl[18][18];
  const int t = threadIdx.x, r = t >> 4, c = t & 15;
  float acc = 0.f;
  for (int ci = 0; ci < Cin; ci++) {
    for (int e = t; e < 324; e += 256) {
      const int rr = e / 18, cc = e % 18;
      float v = 0.f;
      if (rr >= 1 && rr <= 16 && cc >= 1 && cc <= 16)
        v = x[(((size_t)b * Cin + ci) * 16 + rr - 1) * 16 + cc - 1];
      pl[rr][cc] = v;
    }
    __syncthreads();
    const float* wp = wt + ((size_t)co * Cin + ci) * 9;
    acc += pl[r][c]     * wp[0] + pl[r][c + 1]     * wp[1] + pl[r][c + 2]     * wp[2]
         + pl[r + 1][c] * wp[3] + pl[r + 1][c + 1] * wp[4] + pl[r + 1][c + 2] * wp[5]
         + pl[r + 2][c] * wp[6] + pl[r + 2][c + 1] * wp[7] + pl[r + 2][c + 2] * wp[8];
    __syncthreads();
  }
  float v = acc + bias[co];
  v = v * (g[co] * BN_SCALE) + be[co];
  y[(((size_t)b * gridDim.x + co) * 16 + r) * 16 + c] = fmaxf(v, 0.f);
}

// naive GEMM for tiny M; EPI 0=bias 1=bias+relu 2=bias+sigmoid
template<int EPI>
__global__ void small_gemm(const float* __restrict__ W, const float* __restrict__ X,
                           const float* __restrict__ bias, float* __restrict__ out,
                           int M, int N, int K)
{
  const int b = blockIdx.z, m = blockIdx.y;
  const int n = blockIdx.x * 256 + threadIdx.x;
  if (n >= N) return;
  const float* Xb = X + (size_t)b * K * N;
  float a = bias[m];
  for (int k = 0; k < K; k++) a += W[m * K + k] * Xb[(size_t)k * N + n];
  if (EPI == 1) a = fmaxf(a, 0.f);
  if (EPI == 2) a = 1.f / (1.f + __expf(-a));
  out[((size_t)b * M + m) * N + n] = a;
}

__global__ void mul_bc(const float* __restrict__ w, const float* __restrict__ x,
                       float* __restrict__ y, int C, int Npx, size_t total)
{
  const size_t i = (size_t)blockIdx.x * 256 + threadIdx.x;
  if (i >= total) return;
  const int p = (int)(i % Npx);
  const size_t r = i / Npx;
  const int b = (int)(r / C);
  y[i] = w[(size_t)b * Npx + p] * x[i];
}

__global__ void add_inplace(float* __restrict__ y, const float* __restrict__ x, size_t total)
{
  const size_t i = (size_t)blockIdx.x * 256 + threadIdx.x;
  if (i < total) y[i] += x[i];
}

__global__ void sub_to(const float* __restrict__ a, const float* __restrict__ b2,
                       float* __restrict__ o, size_t total)
{
  const size_t i = (size_t)blockIdx.x * 256 + threadIdx.x;
  if (i < total) o[i] = a[i] - b2[i];
}

// XwX accumulation. grid (ceil(N/256), B)
__global__ void __launch_bounds__(256)
xwx_acc(const float* __restrict__ xs, const float* __restrict__ logits,
        float* __restrict__ acc, int N)
{
  const int b = blockIdx.y;
  const int n = blockIdx.x * 256 + threadIdx.x;
  __shared__ float sx[256][9];
  __shared__ float sw[256];
  float X9[9]; float wv = 0.f;
  if (n < N) {
    const float* xp = xs + ((size_t)b * N + n) * 4;
    const float x0 = xp[0], x1 = xp[1], x2 = xp[2], x3 = xp[3];
    X9[0] = x2 * x0; X9[1] = x2 * x1; X9[2] = x2;
    X9[3] = x3 * x0; X9[4] = x3 * x1; X9[5] = x3;
    X9[6] = x0;      X9[7] = x1;      X9[8] = 1.f;
    wv = fmaxf(tanhf(logits[(size_t)b * N + n]), 0.f);
  } else {
    #pragma unroll
    for (int j = 0; j < 9; j++) X9[j] = 0.f;
  }
  #pragma unroll
  for (int j = 0; j < 9; j++) sx[threadIdx.x][j] = X9[j];
  sw[threadIdx.x] = wv;
  __syncthreads();
  if (threadIdx.x < 81) {
    const int ii = threadIdx.x / 9, jj = threadIdx.x % 9;
    float s = 0.f;
    for (int e = 0; e < 256; e++) s += sw[e] * sx[e][ii] * sx[e][jj];
    atomicAdd(acc + b * 81 + threadIdx.x, s);
  }
}

// Jacobi 9x9 symmetric eig; smallest eigenvector, normalized. grid B
__global__ void eigh9_min(const float* __restrict__ acc, float* __restrict__ ehat)
{
  if (threadIdx.x != 0) return;
  const int b = blockIdx.x;
  float A[9][9], V[9][9];
  for (int i = 0; i < 9; i++)
    for (int j = 0; j < 9; j++) {
      A[i][j] = acc[b * 81 + i * 9 + j];
      V[i][j] = (i == j) ? 1.f : 0.f;
    }
  for (int sweep = 0; sweep < 30; sweep++) {
    for (int p = 0; p < 8; p++)
      for (int q = p + 1; q < 9; q++) {
        const float apq = A[p][q];
        if (fabsf(apq) < 1e-12f) continue;
        const float theta = 0.5f * (A[q][q] - A[p][p]) / apq;
        const float tt = ((theta >= 0.f) ? 1.f : -1.f) / (fabsf(theta) + sqrtf(theta * theta + 1.f));
        const float cc = rsqrtf(tt * tt + 1.f);
        const float ss = tt * cc;
        for (int k = 0; k < 9; k++) {
          const float akp = A[k][p], akq = A[k][q];
          A[k][p] = cc * akp - ss * akq;
          A[k][q] = ss * akp + cc * akq;
        }
        for (int k = 0; k < 9; k++) {
          const float apk = A[p][k], aqk = A[q][k];
          A[p][k] = cc * apk - ss * aqk;
          A[q][k] = ss * apk + cc * aqk;
        }
        for (int k = 0; k < 9; k++) {
          const float vkp = V[k][p], vkq = V[k][q];
          V[k][p] = cc * vkp - ss * vkq;
          V[k][q] = ss * vkp + cc * vkq;
        }
      }
  }
  int kmin = 0; float emin = A[0][0];
  for (int k = 1; k < 9; k++) if (A[k][k] < emin) { emin = A[k][k]; kmin = k; }
  float nrm = 0.f;
  for (int k = 0; k < 9; k++) nrm += V[k][kmin] * V[k][kmin];
  nrm = rsqrtf(nrm + 1e-30f);
  for (int k = 0; k < 9; k++) ehat[b * 9 + k] = V[k][kmin] * nrm;
}

// ---------------------------------------------------------------------------
// Parameter index map (dict insertion order)
// ---------------------------------------------------------------------------
enum {
  IN_XS = 0, IN_D = 1, IN_GPE = 2,
  AL_QW = 3, AL_QB, AL_KW, AL_KB, AL_VW, AL_VB, AL_MHW, AL_MHB,
  AL_C1W, AL_C1B, AL_C1G, AL_C1BE, AL_C2W, AL_C2B,
  DE_QW, DE_QB, DE_KW, DE_KB, DE_VW, DE_VB, DE_MHW, DE_MHB,
  DE_C1W, DE_C1B, DE_C1G, DE_C1BE, DE_C2W, DE_C2B,
  G_W, G_B,
  KC1W, KC1B, KC1G, KC1BE, KC2W, KC2B,
  KM1W, KM1B, KM1G, KM1BE, KM2W, KM2B,
  LO_W, LO_B,
  RB_BASE,                  // 47: per-block 12 entries x 3 blocks
  SC_W = 83, SC_B,
  I1W = 85, I1B, I1G, I1BE,
  I2W, I2B, I2G, I2BE,
  I3W, I3B, I3G, I3BE,
  I4W = 97, I4B,
  TC1W = 99, TC1B, TCG, TCBE, TC2W = 103, TC2B
};

extern "C" void kernel_launch(void* const* d_in, const int* in_sizes, int n_in,
                              void* d_out, int out_size, void* d_ws, size_t ws_size,
                              hipStream_t stream)
{
  const int B = 4, C = 128, N = 8000, G = 256;
  const int N8 = N * 8;
  auto F = [&](int i) -> const float* { return (const float*)d_in[i]; };
  const float* xs  = F(IN_XS);
  const float* din = F(IN_D);
  const float* gpe = F(IN_GPE);

  char* wp = (char*)d_ws;
  auto carve = [&](size_t bytes) -> char* {
    char* r = wp; wp += (bytes + 255) & ~(size_t)255; return r;
  };
  auto cvt = [&](const float* src, _Float16* dst, size_t n) {
    f32_to_f16_k<<<dim3((unsigned)((n + 255) / 256)), 256, 0, stream>>>(src, dst, n);
  };

  // --- f16 weight copies (converted once, reused by all tiles/batches) ---
  static const int wlist[21] = {
    AL_QW, AL_KW, AL_VW, AL_MHW, AL_C1W, AL_C2W,
    DE_QW, DE_KW, DE_VW, DE_MHW, DE_C1W, DE_C2W,
    KC1W, KC2W, KM1W, KM2W, LO_W, SC_W, I1W, TC1W, TC2W };
  _Float16* WH[128] = {};
  for (int ii = 0; ii < 21; ii++) {
    const int pi = wlist[ii];
    const size_t n = (size_t)in_sizes[pi];
    WH[pi] = (_Float16*)carve(2 * n);
    cvt(F(pi), WH[pi], n);
  }

  float*     sq    = (float*)carve(sizeof(float) * B * N);
  int*       idx   = (int*)  carve(sizeof(int)   * B * N * 8);
  _Float16*  dh16  = (_Float16*)carve(2ull * B * 128 * N);
  _Float16*  gpeh  = (_Float16*)carve(2ull * B * 128 * G);
  _Float16*  gf    = (_Float16*)carve(2ull * B * 256 * N8);
  _Float16*  bufA  = (_Float16*)carve(2ull * B * 256 * N8);
  _Float16*  bufB  = (_Float16*)carve(2ull * B * 128 * N8);
  _Float16*  bufC  = gf; // gf dead after kc1; reuse for km1 output
  _Float16*  mxk   = (_Float16*)carve(2ull * B * 128 * N);
  float*     d_loc = (float*)carve(4ull * B * 128 * N);
  _Float16*  cat   = (_Float16*)carve(2ull * B * 256 * N);
  _Float16*  t1    = (_Float16*)carve(2ull * B * 256 * N);
  float*     dU    = (float*)carve(4ull * B * 128 * N);
  _Float16*  dUh   = (_Float16*)carve(2ull * B * 128 * N);
  _Float16*  qA    = (_Float16*)carve(2ull * B * 128 * G);
  _Float16*  kA    = (_Float16*)carve(2ull * B * 128 * N);
  _Float16*  vA    = (_Float16*)carve(2ull * B * 128 * N);
  _Float16*  attA  = (_Float16*)carve(2ull * B * 128 * G);
  _Float16*  mhA   = (_Float16*)carve(2ull * B * 128 * G);
  _Float16*  catA  = (_Float16*)carve(2ull * B * 256 * G);
  _Float16*  t2    = (_Float16*)carve(2ull * B * 256 * G);
  float*     gridd = (float*)carve(4ull * B * 128 * G);
  _Float16*  griddh= (_Float16*)carve(2ull * B * 128 * G);
  float*     r1    = (float*)carve(4ull * B * 64 * G);
  float*     wgt   = (float*)carve(4ull * B * G);
  float*     wx    = (float*)carve(4ull * B * 128 * G);
  float*     r2    = (float*)carve(4ull * B * 128 * G);
  float*     r3    = (float*)carve(4ull * B * 128 * G);
  float*     gridd2= (float*)carve(4ull * B * 128 * G);
  _Float16*  g2h   = (_Float16*)carve(2ull * B * 128 * G);
  _Float16*  qD    = (_Float16*)carve(2ull * B * 128 * N);
  _Float16*  kD    = (_Float16*)carve(2ull * B * 128 * G);
  _Float16*  vD    = (_Float16*)carve(2ull * B * 128 * G);
  _Float16*  attD  = (_Float16*)carve(2ull * B * 128 * N);
  _Float16*  mhD   = (_Float16*)carve(2ull * B * 128 * N);
  _Float16*  catD  = (_Float16*)carve(2ull * B * 256 * N);
  _Float16*  t3    = (_Float16*)carve(2ull * B * 256 * N);
  float*     diff  = (float*)carve(4ull * B * 128 * N);
  _Float16*  diffh = (_Float16*)carve(2ull * B * 128 * N);
  float*     z1    = (float*)carve(4ull * B * 64 * N);
  float*     z2    = (float*)carve(4ull * B * 16 * N);
  float*     z3    = (float*)carve(4ull * B * 4 * N);
  float*     xacc  = (float*)carve(sizeof(float) * B * 81);

  float* out_d     = (float*)d_out;
  float* out_logit = out_d + (size_t)B * C * N;
  float* out_e     = out_logit + (size_t)B * N;

  cvt(din, dh16, (size_t)B * 128 * N);
  cvt(gpe, gpeh, (size_t)B * 128 * G);

  // ----- Stage 1: lcl(d) -----
  sqnorm_k<<<dim3((B * N + 255) / 256), 256, 0, stream>>>(din, sq, N, B * N);
  knn_top8<<<dim3(N / 64, B), 256, 0, stream>>>(dh16, sq, idx, N);
  build_gf<<<dim3(N / 32, B), 256, 0, stream>>>(din, idx, F(G_W), F(G_B), gf, N);

  gemm_wmma<_Float16, 7><<<dim3(N8 / 64, 256 / 64, B), 256, 0, stream>>>(
      WH[KC1W], gf, F(KC1B), F(KC1G), F(KC1BE), nullptr, bufA, 256, N8, 256);
  gemm_wmma<_Float16, 1><<<dim3(N8 / 64, 128 / 64, B), 256, 0, stream>>>(
      WH[KC2W], bufA, F(KC2B), nullptr, nullptr, nullptr, bufB, 128, N8, 256);
  gemm_wmma<_Float16, 7><<<dim3(N8 / 64, 128 / 64, B), 256, 0, stream>>>(
      WH[KM1W], bufB, F(KM1B), F(KM1G), F(KM1BE), nullptr, bufC, 128, N8, 128);
  gemm_wmma<_Float16, 1><<<dim3(N8 / 64, 128 / 64, B), 256, 0, stream>>>(
      WH[KM2W], bufC, F(KM2B), nullptr, nullptr, nullptr, bufB, 128, N8, 128);
  { size_t tot = (size_t)B * 128 * N;
    max_over_k<<<dim3((unsigned)((tot + 255) / 256)), 256, 0, stream>>>(bufB, mxk, tot); }
  gemm_wmma<float, 1><<<dim3(N / 64, 128 / 64, B), 256, 0, stream>>>(
      WH[LO_W], mxk, F(LO_B), nullptr, nullptr, nullptr, d_loc, 128, N, 128);

  // ----- Stage 2: tc path, d -> dU -----
  { size_t tot = (size_t)B * 256 * N;
    concat2_f16<float, float><<<dim3((unsigned)((tot + 255) / 256)), 256, 0, stream>>>(
        din, d_loc, cat, C, N, tot); }
  gemm_wmma<_Float16, 7><<<dim3(N / 64, 256 / 64, B), 256, 0, stream>>>(
      WH[TC1W], cat, F(TC1B), F(TCG), F(TCBE), nullptr, t1, 256, N, 256);
  gemm_wmma<float, 9><<<dim3(N / 64, 128 / 64, B), 256, 0, stream>>>(
      WH[TC2W], t1, F(TC2B), nullptr, nullptr, din, dU, 128, N, 256);
  cvt(dU, dUh, (size_t)B * 128 * N);

  // ----- Stage 3: attn 'al' (gpe queries, dU keys) -----
  gemm_wmma<_Float16, 1><<<dim3(G / 64, 128 / 64, B), 256, 0, stream>>>(
      WH[AL_QW], gpeh, F(AL_QB), nullptr, nullptr, nullptr, qA, 128, G, 128);
  gemm_wmma<_Float16, 1><<<dim3(N / 64, 128 / 64, B), 256, 0, stream>>>(
      WH[AL_KW], dUh, F(AL_KB), nullptr, nullptr, nullptr, kA, 128, N, 128);
  gemm_wmma<_Float16, 1><<<dim3(N / 64, 128 / 64, B), 256, 0, stream>>>(
      WH[AL_VW], dUh, F(AL_VB), nullptr, nullptr, nullptr, vA, 128, N, 128);
  attn_flash<<<dim3(G / 64, 4, B), 128, 0, stream>>>(qA, kA, vA, attA, G, N);
  gemm_wmma<_Float16, 1><<<dim3(G / 64, 128 / 64, B), 256, 0, stream>>>(
      WH[AL_MHW], attA, F(AL_MHB), nullptr, nullptr, nullptr, mhA, 128, G, 128);
  { size_t tot = (size_t)B * 256 * G;
    concat2_f16<float, _Float16><<<dim3((unsigned)((tot + 255) / 256)), 256, 0, stream>>>(
        gpe, mhA, catA, C, G, tot); }
  gemm_wmma<_Float16, 7><<<dim3(G / 64, 256 / 64, B), 256, 0, stream>>>(
      WH[AL_C1W], catA, F(AL_C1B), F(AL_C1G), F(AL_C1BE), nullptr, t2, 256, G, 256);
  gemm_wmma<float, 9><<<dim3(G / 64, 128 / 64, B), 256, 0, stream>>>(
      WH[AL_C2W], t2, F(AL_C2B), nullptr, nullptr, gpe, gridd, 128, G, 256);

  // ----- Stage 4: filter_net -----
  for (int rb = 0; rb < 3; rb++) {
    const int pb = RB_BASE + rb * 12;
    small_gemm<1><<<dim3(1, 64, B), 256, 0, stream>>>(
        F(pb + 8), gridd, F(pb + 9), r1, 64, G, 128);
    small_gemm<2><<<dim3(1, 1, B), 256, 0, stream>>>(
        F(pb + 10), r1, F(pb + 11), wgt, 1, G, 64);
    { size_t tot = (size_t)B * 128 * G;
      mul_bc<<<dim3((unsigned)((tot + 255) / 256)), 256, 0, stream>>>(wgt, gridd, wx, 128, G, tot); }
    conv3x3_bnrelu<<<dim3(128, B), 256, 0, stream>>>(
        wx, F(pb + 0), F(pb + 1), F(pb + 2), F(pb + 3), r2, 128);
    conv3x3_bnrelu<<<dim3(128, B), 256, 0, stream>>>(
        r2, F(pb + 4), F(pb + 5), F(pb + 6), F(pb + 7), r3, 128);
    { size_t tot = (size_t)B * 128 * G;
      add_inplace<<<dim3((unsigned)((tot + 255) / 256)), 256, 0, stream>>>(gridd, r3, tot); }
  }
  cvt(gridd, griddh, (size_t)B * 128 * G);
  gemm_wmma<float, 1><<<dim3(G / 64, 128 / 64, B), 256, 0, stream>>>(
      WH[SC_W], griddh, F(SC_B), nullptr, nullptr, nullptr, gridd2, 128, G, 128);
  cvt(gridd2, g2h, (size_t)B * 128 * G);

  // ----- Stage 5: attn 'de' (dU queries, grid keys) -> d_new (into d_out) -----
  gemm_wmma<_Float16, 1><<<dim3(N / 64, 128 / 64, B), 256, 0, stream>>>(
      WH[DE_QW], dUh, F(DE_QB), nullptr, nullptr, nullptr, qD, 128, N, 128);
  gemm_wmma<_Float16, 1><<<dim3(G / 64, 128 / 64, B), 256, 0, stream>>>(
      WH[DE_KW], g2h, F(DE_KB), nullptr, nullptr, nullptr, kD, 128, G, 128);
  gemm_wmma<_Float16, 1><<<dim3(G / 64, 128 / 64, B), 256, 0, stream>>>(
      WH[DE_VW], g2h, F(DE_VB), nullptr, nullptr, nullptr, vD, 128, G, 128);
  attn_flash<<<dim3(N / 64, 4, B), 128, 0, stream>>>(qD, kD, vD, attD, N, G);
  gemm_wmma<_Float16, 1><<<dim3(N / 64, 128 / 64, B), 256, 0, stream>>>(
      WH[DE_MHW], attD, F(DE_MHB), nullptr, nullptr, nullptr, mhD, 128, N, 128);
  { size_t tot = (size_t)B * 256 * N;
    concat2_f16<float, _Float16><<<dim3((unsigned)((tot + 255) / 256)), 256, 0, stream>>>(
        dU, mhD, catD, C, N, tot); }
  gemm_wmma<_Float16, 7><<<dim3(N / 64, 256 / 64, B), 256, 0, stream>>>(
      WH[DE_C1W], catD, F(DE_C1B), F(DE_C1G), F(DE_C1BE), nullptr, t3, 256, N, 256);
  gemm_wmma<float, 9><<<dim3(N / 64, 128 / 64, B), 256, 0, stream>>>(
      WH[DE_C2W], t3, F(DE_C2B), nullptr, nullptr, dU, out_d, 128, N, 256);

  // ----- Stage 6: inlier_pre(d_new - dU) -> logits -----
  { size_t tot = (size_t)B * 128 * N;
    sub_to<<<dim3((unsigned)((tot + 255) / 256)), 256, 0, stream>>>(out_d, dU, diff, tot); }
  cvt(diff, diffh, (size_t)B * 128 * N);
  gemm_wmma<float, 1><<<dim3(N / 64, 64 / 64, B), 256, 0, stream>>>(
      WH[I1W], diffh, F(I1B), nullptr, nullptr, nullptr, z1, 64, N, 128);
  inorm_bn_relu_k<<<dim3(B * 64), 256, 0, stream>>>(z1, F(I1G), F(I1BE), 64, N);
  small_gemm<0><<<dim3((N + 255) / 256, 16, B), 256, 0, stream>>>(F(I2W), z1, F(I2B), z2, 16, N, 64);
  inorm_bn_relu_k<<<dim3(B * 16), 256, 0, stream>>>(z2, F(I2G), F(I2BE), 16, N);
  small_gemm<0><<<dim3((N + 255) / 256, 4, B), 256, 0, stream>>>(F(I3W), z2, F(I3B), z3, 4, N, 16);
  inorm_bn_relu_k<<<dim3(B * 4), 256, 0, stream>>>(z3, F(I3G), F(I3BE), 4, N);
  small_gemm<0><<<dim3((N + 255) / 256, 1, B), 256, 0, stream>>>(F(I4W), z3, F(I4B), out_logit, 1, N, 4);

  // ----- Stage 7: weighted 8-points -----
  hipMemsetAsync(xacc, 0, sizeof(float) * B * 81, stream);
  xwx_acc<<<dim3((N + 255) / 256, B), 256, 0, stream>>>(xs, out_logit, xacc, N);
  eigh9_min<<<dim3(B), 32, 0, stream>>>(xacc, out_e);
}
